// KNNClassifier_25116968747349
// MI455X (gfx1250) — matrix operand order repaired and checked by reading, not compiled, not run on hardware
//
#include <hip/hip_runtime.h>

// ---------------------------------------------------------------------------
// KNN classifier, MI455X (gfx1250): f16 WMMA GEMM fused with top-k selection.
//   Q=4096 queries, N=100000 train rows, D=512, k=16, 100 classes.
//   Each wave: 16 queries x 64 train cols strip, 4 f32 accumulators,
//   A-fragment loaded once per K-step -> 2.5 ds_load_b128 per WMMA.
//   B tiles staged LDS-direct by the Tensor Data Mover (tensor_load_to_lds,
//   TENSORcnt), bypassing VGPRs entirely.
// ---------------------------------------------------------------------------

#define NQ      4096
#define DIM     512
#define NTRAIN  100000
#define NCLASS  100
#define KSEL    16
#define QT      64        // queries per block (4 waves x 16)
#define NSPLIT  32        // train-dimension grid split
#define NCHUNK  3136      // rows per chunk = 49*64
#define NPAD    (NSPLIT * NCHUNK)   // 100352 rows incl. zero padding
#define NT      64        // train rows staged in LDS per iteration

typedef __attribute__((ext_vector_type(16))) _Float16 v16h;
typedef __attribute__((ext_vector_type(8)))  _Float16 v8h;
typedef __attribute__((ext_vector_type(8)))  float    v8f;
typedef __attribute__((ext_vector_type(4)))  _Float16 h4;
typedef __attribute__((ext_vector_type(4)))  unsigned int v4u;
typedef __attribute__((ext_vector_type(8)))  unsigned int v8u;

// ---------------------------------------------------------------------------
// f32 -> f16 conversion + squared row norms. One wave (32 lanes) per row.
// ---------------------------------------------------------------------------
__global__ void __launch_bounds__(256) prep_f16(const float* __restrict__ src,
                                                _Float16* __restrict__ dst,
                                                float* __restrict__ n2,
                                                int rows) {
  const int wave = threadIdx.x >> 5;
  const int lane = threadIdx.x & 31;
  const int r = blockIdx.x * 8 + wave;
  if (r >= rows) return;
  const float* s = src + (size_t)r * DIM;
  _Float16*    d = dst + (size_t)r * DIM;
  float acc = 0.0f;
#pragma unroll
  for (int c = 0; c < 4; ++c) {
    const int idx = c * 128 + lane * 4;
    float4 v = *(const float4*)(s + idx);
    acc += v.x * v.x + v.y * v.y + v.z * v.z + v.w * v.w;
    h4 h;
    h.x = (_Float16)v.x; h.y = (_Float16)v.y;
    h.z = (_Float16)v.z; h.w = (_Float16)v.w;
    *(h4*)(d + idx) = h;
  }
#pragma unroll
  for (int off = 16; off; off >>= 1) acc += __shfl_xor(acc, off, 32);
  if (lane == 0) n2[r] = acc;
}

// Zero-fill the padded tail rows of T16 (rows NTRAIN..NPAD-1), every launch.
__global__ void __launch_bounds__(256) zero_pad(_Float16* __restrict__ dst) {
  const int total = (NPAD - NTRAIN) * DIM / 8;  // uint4 count
  uint4* d = (uint4*)(dst + (size_t)NTRAIN * DIM);
  const uint4 z = {0u, 0u, 0u, 0u};
  for (int i = blockIdx.x * 256 + threadIdx.x; i < total; i += gridDim.x * 256)
    d[i] = z;
}

// ---------------------------------------------------------------------------
// Fused WMMA GEMM + running top-16 per query over one chunk of train rows.
// grid = (NQ/QT, NSPLIT), block = 128 (4 waves; wave w owns queries w*16..+15)
// ---------------------------------------------------------------------------
__global__ void __launch_bounds__(128) knn_chunk(
    const _Float16* __restrict__ X16, const _Float16* __restrict__ T16,
    const float* __restrict__ x2g, const float* __restrict__ t2g,
    float* __restrict__ cand_d, int* __restrict__ cand_i) {
  __shared__ __align__(32) _Float16 Atile[QT * DIM];   // 64 KB
  __shared__ __align__(32) _Float16 Btile[NT * DIM];   // 64 KB
  __shared__ float topd[QT * KSEL];
  __shared__ int   topi[QT * KSEL];
  __shared__ float x2s[QT];
  __shared__ float t2s[NT];
  __shared__ float d2t[4][16 * 17];                    // padded score spill

  const int tid  = threadIdx.x;
  const int wave = tid >> 5;
  const int lane = tid & 31;
  const int q0   = blockIdx.x * QT;
  const int n0   = blockIdx.y * NCHUNK;
  const int m0   = wave * 16;

  // Stage A (64 query rows are contiguous in X16) + init top-k lists.
  {
    const uint4* srcv = (const uint4*)(X16 + (size_t)q0 * DIM);
    uint4* dstv = (uint4*)Atile;
    for (int i = tid; i < QT * DIM / 8; i += 128) dstv[i] = srcv[i];
  }
  for (int i = tid; i < QT * KSEL; i += 128) { topd[i] = 3.0e38f; topi[i] = 0; }
  if (tid < QT) x2s[tid] = x2g[q0 + tid];
  __syncthreads();

  // Per-16x16-tile selection epilogue: threshold test in registers, rare
  // LDS spill + serial insertion only when a candidate beats the 16th-best.
  auto select_tile = [&](v8f c, int nt, int nbase) {
    const float tn = t2s[nt * 16 + (lane & 15)];
    float d2v[8];
    int hit = 0;
#pragma unroll
    for (int r = 0; r < 8; ++r) {
      const int m = r + (lane >> 4) * 8;      // C layout: M = vgpr(+8), N = lane%16
      const float d2 = x2s[m0 + m] + tn - 2.0f * c[r];
      d2v[r] = d2;
      hit |= (d2 < topd[(m0 + m) * KSEL + KSEL - 1]) ? 1 : 0;
    }
    if (__any(hit)) {
#pragma unroll
      for (int r = 0; r < 8; ++r) {
        const int m = r + (lane >> 4) * 8;
        d2t[wave][m * 17 + (lane & 15)] = d2v[r];
      }
      asm volatile("s_wait_dscnt 0" ::: "memory");
      if (lane < 16) {  // lane = query row within this wave's M tile
        const int qi = m0 + lane;
        float* td = &topd[qi * KSEL];
        int*   ti = &topi[qi * KSEL];
        for (int j = 0; j < 16; ++j) {
          const int colg = nbase + nt * 16 + j;
          if (colg >= NTRAIN) break;  // columns ascend; rest invalid too
          const float d2 = d2t[wave][lane * 17 + j];
          if (d2 < td[KSEL - 1]) {
            int p = KSEL - 1;
            while (p > 0 && td[p - 1] > d2) {  // strict: stable for ties
              td[p] = td[p - 1]; ti[p] = ti[p - 1]; --p;
            }
            td[p] = d2; ti[p] = colg;
          }
        }
      }
      asm volatile("s_wait_dscnt 0" ::: "memory");
    }
  };

  // Tensor DMA descriptor invariants (ISA cdna5 8.3/8.4): 2D tensor, 8-byte
  // elements; tile = 64 rows x 128 units (1024 B); row stride = 128 units.
  const unsigned lds_b = (unsigned)(uintptr_t)&Btile[0];  // generic low 32b = LDS offset

  for (int nb = 0; nb < NCHUNK / NT; ++nb) {
    const int nbase = n0 + nb * NT;
    __syncthreads();  // previous Btile readers done
    if (wave == 0) {
      const unsigned long long gaddr =
          (unsigned long long)(uintptr_t)T16 + (unsigned long long)nbase * (DIM * 2);
      v4u g0;
      g0[0] = 1u;                                   // count=1 valid descriptor
      g0[1] = lds_b;                                // lds_addr
      g0[2] = (unsigned)(gaddr & 0xffffffffu);      // global_addr[31:0]
      g0[3] = (unsigned)((gaddr >> 32) & 0x01ffffffu) | 0x80000000u;  // [56:32] | type=2
      v8u g1;
      g1[0] = 0x00030000u;                          // wg_mask=0, data_size=3 (8B)
      g1[1] = (128u & 0xffffu) << 16;               // tensor_dim0[15:0]=128
      g1[2] = (128u >> 16) | ((NPAD & 0xffffu) << 16);   // dim0 hi | tensor_dim1 lo
      g1[3] = ((unsigned)NPAD >> 16) | (128u << 16);     // tensor_dim1 hi | tile_dim0=128
      g1[4] = (unsigned)NT;                         // tile_dim1=64, tile_dim2=0
      g1[5] = 128u;                                 // tensor_dim0_stride[31:0]=128
      g1[6] = 0u;                                   // stride hi | tensor_dim1_stride lo
      g1[7] = 0u;                                   // tensor_dim1_stride hi
      asm volatile("tensor_load_to_lds %0, %1" :: "s"(g0), "s"(g1) : "memory");
      __builtin_amdgcn_s_wait_tensorcnt(0);
    }
    if (tid < NT) t2s[tid] = (nbase + tid < NTRAIN) ? t2g[nbase + tid] : 0.0f;
    __syncthreads();

    // 16q x 64n strip: A-fragment loaded once, 4 B-fragments + 4 WMMAs per K.
    v8f c0 = {0.f, 0.f, 0.f, 0.f, 0.f, 0.f, 0.f, 0.f};
    v8f c1 = c0, c2 = c0, c3 = c0;
    const int arow  = lane & 15;
    const int ahalf = lane >> 4;  // lane half h holds K 16h..16h+15 (contiguous; B same)
    const _Float16* abase = Atile + (m0 + arow) * DIM;
    const _Float16* bbase = Btile + arow * DIM;  // B column = train row
#pragma unroll
    for (int ks = 0; ks < DIM / 32; ++ks) {
      const int k0 = ks * 32;
      union { v16h v; v8h h[2]; } a;
      a.v = *(const v16h*)(abase + k0 + ahalf * 16);         // K k0..+15 / k0+16..+31
      v16h b0 = *(const v16h*)(bbase + 0 * 16 * DIM + k0 + ahalf * 16);
      v16h b1 = *(const v16h*)(bbase + 1 * 16 * DIM + k0 + ahalf * 16);
      v16h b2 = *(const v16h*)(bbase + 2 * 16 * DIM + k0 + ahalf * 16);
      v16h b3 = *(const v16h*)(bbase + 3 * 16 * DIM + k0 + ahalf * 16);
      c0 = __builtin_amdgcn_wmma_f32_16x16x32_f16(false, a.v, false, b0,
                                                  (short)0, c0, false, false);
      c1 = __builtin_amdgcn_wmma_f32_16x16x32_f16(false, a.v, false, b1,
                                                  (short)0, c1, false, false);
      c2 = __builtin_amdgcn_wmma_f32_16x16x32_f16(false, a.v, false, b2,
                                                  (short)0, c2, false, false);
      c3 = __builtin_amdgcn_wmma_f32_16x16x32_f16(false, a.v, false, b3,
                                                  (short)0, c3, false, false);
    }
    select_tile(c0, 0, nbase);
    select_tile(c1, 1, nbase);
    select_tile(c2, 2, nbase);
    select_tile(c3, 3, nbase);
  }
  __syncthreads();
  if (lane < 16) {
    const int qi = m0 + lane;
    const size_t base = ((size_t)(q0 + qi) * NSPLIT + blockIdx.y) * KSEL;
#pragma unroll
    for (int j = 0; j < KSEL; ++j) {
      cand_d[base + j] = topd[qi * KSEL + j];
      cand_i[base + j] = topi[qi * KSEL + j];
    }
  }
}

// ---------------------------------------------------------------------------
// Merge 32 sorted candidate lists per query, vote over 100 classes.
// One wave per query; lane l owns one sorted 16-list (head-pointer merge).
// ---------------------------------------------------------------------------
__global__ void __launch_bounds__(128) knn_merge(
    const float* __restrict__ cand_d, const int* __restrict__ cand_i,
    const int* __restrict__ y_train, float* __restrict__ out) {
  const int wave = threadIdx.x >> 5;
  const int lane = threadIdx.x & 31;
  const int q = blockIdx.x * 4 + wave;
  const size_t base = (size_t)q * (NSPLIT * KSEL) + (size_t)lane * KSEL;
  int head = 0;
  int c0 = 0, c1 = 0, c2 = 0, c3 = 0;
  for (int t = 0; t < KSEL; ++t) {
    const float d = (head < KSEL) ? cand_d[base + head] : 3.0e38f;
    const int   i = (head < KSEL) ? cand_i[base + head] : 0x7fffffff;
    float bd = d; int bi = i;
#pragma unroll
    for (int off = 16; off; off >>= 1) {
      const float od = __shfl_xor(bd, off, 32);
      const int   oi = __shfl_xor(bi, off, 32);
      if (od < bd || (od == bd && oi < bi)) { bd = od; bi = oi; }
    }
    if (d == bd && i == bi) ++head;          // winner advances its list
    const int lab = y_train[bi];             // same addr all lanes -> broadcast
    c0 += (lab == lane);
    c1 += (lab == lane + 32);
    c2 += (lab == lane + 64);
    c3 += (lane < 4 && lab == lane + 96);
  }
  // argmax count, tie -> smallest class (matches jnp.argmax first-max).
  int key = (c0 << 7) | (127 - lane);
  int k1  = (c1 << 7) | (127 - (lane + 32)); if (k1 > key) key = k1;
  int k2  = (c2 << 7) | (127 - (lane + 64)); if (k2 > key) key = k2;
  if (lane < 4) { int k3 = (c3 << 7) | (127 - (lane + 96)); if (k3 > key) key = k3; }
#pragma unroll
  for (int off = 16; off; off >>= 1) {
    const int ok = __shfl_xor(key, off, 32);
    if (ok > key) key = ok;
  }
  if (lane == 0) out[q] = (float)(127 - (key & 127));
}

// ---------------------------------------------------------------------------
extern "C" void kernel_launch(void* const* d_in, const int* in_sizes, int n_in,
                              void* d_out, int out_size, void* d_ws, size_t ws_size,
                              hipStream_t stream) {
  (void)in_sizes; (void)n_in; (void)out_size; (void)ws_size;
  const float* X  = (const float*)d_in[0];
  const float* Xt = (const float*)d_in[1];
  const int*   y  = (const int*)d_in[2];
  // d_in[3] = k (device scalar); reference fixes k=16, baked in as KSEL.

  char* ws = (char*)d_ws;
  size_t off = 0;
  auto carve = [&](size_t bytes) -> void* {
    void* p = ws + off;
    off = (off + bytes + 255) & ~(size_t)255;
    return p;
  };
  _Float16* X16 = (_Float16*)carve((size_t)NQ * DIM * 2);
  _Float16* T16 = (_Float16*)carve((size_t)NPAD * DIM * 2);
  float* x2 = (float*)carve((size_t)NQ * 4);
  float* t2 = (float*)carve((size_t)NTRAIN * 4);
  float* cd = (float*)carve((size_t)NQ * NSPLIT * KSEL * 4);
  int*   ci = (int*)carve((size_t)NQ * NSPLIT * KSEL * 4);

  prep_f16<<<NQ / 8, 256, 0, stream>>>(X, X16, x2, NQ);
  prep_f16<<<(NTRAIN + 7) / 8, 256, 0, stream>>>(Xt, T16, t2, NTRAIN);
  zero_pad<<<88, 256, 0, stream>>>(T16);

  dim3 grid(NQ / QT, NSPLIT);
  knn_chunk<<<grid, 128, 0, stream>>>(X16, T16, x2, t2, cd, ci);

  knn_merge<<<NQ / 4, 128, 0, stream>>>(cd, ci, y, (float*)d_out);
}
